// MedDiffusionModel_27255862461113
// MI455X (gfx1250) — matrix operand order repaired and checked by reading, not compiled, not run on hardware
//
#include <hip/hip_runtime.h>
#include <hip/hip_bf16.h>

typedef __attribute__((ext_vector_type(16))) _Float16 v16h;
typedef __attribute__((ext_vector_type(8)))  _Float16 v8h;
typedef __attribute__((ext_vector_type(8)))  float    v8f;

#define B_ 128
#define L_ 64
#define V_ 8192
#define E_ 128
#define H_ 128
#define STEPS_ 10
#define ROWS_ (B_ * L_)   // 8192
#define BETA0_ 1.0e-4f
#define BETA1_ 2.0e-2f

__device__ __forceinline__ float beta_at(int i) {
  return BETA0_ + (BETA1_ - BETA0_) * ((float)i / 9.0f);
}

__device__ __forceinline__ float sigmoidf_(float x) {
  return 1.0f / (1.0f + __expf(-x));
}

// Assemble A fragment (contiguous k order, matching B) from f16 memory:
// halves 0..15 = K[ka..ka+15]; p points at k0+ka.
__device__ __forceinline__ v16h load_a16(const _Float16* __restrict__ p) {
  v8h lo = *(const v8h*)(p);
  v8h hi = *(const v8h*)(p + 8);
  v16h a;
#pragma unroll
  for (int i = 0; i < 8; ++i) { a[i] = lo[i]; a[i + 8] = hi[i]; }
  return a;
}

// ---------------------------------------------------------------------------
// f32 -> f16 conversion (weights / staging)
// ---------------------------------------------------------------------------
__global__ void cvt16_kernel(const float* __restrict__ src,
                             _Float16* __restrict__ dst, int n)
{
  int i = blockIdx.x * blockDim.x + threadIdx.x;
  if (i < n) dst[i] = (_Float16)src[i];
}

// ---------------------------------------------------------------------------
// v = (padded_x @ W_visit.T + bin_embed[delta]) * mask     [8192 x 128]
// A stays f32 (binary, streamed once from HBM: 256MB @ 23.3TB/s is the floor);
// B is pre-converted f16 (aligned 32B vector loads).  Writes v (f32) + v_h (f16).
// ---------------------------------------------------------------------------
__global__ void visit_embed_kernel(const float* __restrict__ padded_x,
                                   const _Float16* __restrict__ Wv_h,
                                   const int*   __restrict__ padded_bins,
                                   const float* __restrict__ visit_mask,
                                   const float* __restrict__ bin_embed,
                                   float* __restrict__ v,
                                   _Float16* __restrict__ v_h)
{
  int w = blockIdx.x * (blockDim.x >> 5) + (threadIdx.x >> 5);
  if (w >= 512 * 8) return;
  int nt = w & 7, mt = w >> 3;
  int lane = threadIdx.x & 31;
  int r = lane & 15, hi = lane >> 4;
  const float*    arow = padded_x + (size_t)(mt * 16 + r) * V_;
  const _Float16* brow = Wv_h + (size_t)(nt * 16 + r) * V_;
  const int ka = hi * 16, kb = hi * 16;
  v8f acc = {};
  for (int k0 = 0; k0 < V_; k0 += 32) {
    if (k0 + 256 < V_) {
      __builtin_prefetch((const void*)(arow + k0 + 256), 0, 0); // global_prefetch_b8
    }
    v16h a;
#pragma unroll
    for (int i = 0; i < 8; ++i) {
      a[i]     = (_Float16)arow[k0 + ka + i];
      a[i + 8] = (_Float16)arow[k0 + ka + 8 + i];
    }
    v16h b = *(const v16h*)(brow + k0 + kb);
    acc = __builtin_amdgcn_wmma_f32_16x16x32_f16(false, a, false, b,
                                                 (short)0, acc, false, false);
  }
  int n = nt * 16 + r;
#pragma unroll
  for (int i = 0; i < 8; ++i) {
    int m = mt * 16 + hi * 8 + i;
    int l = m & (L_ - 1);
    float mask = visit_mask[m];
    int delta = 0;
    if (l > 0 && mask > 0.0f) {
      int d = padded_bins[m] - padded_bins[m - 1];
      delta = d < 0 ? 0 : (d > 513 ? 513 : d);
    }
    float val = (acc[i] + bin_embed[(size_t)delta * E_ + n]) * mask;
    v[(size_t)m * E_ + n]   = val;
    v_h[(size_t)m * E_ + n] = (_Float16)val;
  }
}

// ---------------------------------------------------------------------------
// Generic C[M x N] = A16[M x K] @ Wt16[N x K]^T (+bias1+bias2)(relu)
// Dual output: f32 (Cf) and/or f16 (Ch); either may be null.
// ---------------------------------------------------------------------------
__global__ void gemm_h_kernel(const _Float16* __restrict__ A, int lda,
                              const _Float16* __restrict__ Wt, int K,
                              float* __restrict__ Cf, _Float16* __restrict__ Ch,
                              int ldc, int Mtiles, int Ntiles,
                              const float* __restrict__ bias1,
                              const float* __restrict__ bias2,
                              int relu)
{
  int w = blockIdx.x * (blockDim.x >> 5) + (threadIdx.x >> 5);
  if (w >= Mtiles * Ntiles) return;
  int nt = w % Ntiles, mt = w / Ntiles;
  int lane = threadIdx.x & 31;
  int r = lane & 15, hi = lane >> 4;
  const _Float16* arow = A + (size_t)(mt * 16 + r) * lda;
  const _Float16* brow = Wt + (size_t)(nt * 16 + r) * K;
  const int ka = hi * 16, kb = hi * 16;
  v8f acc = {};
  for (int k0 = 0; k0 < K; k0 += 32) {
    v16h a = load_a16(arow + k0 + ka);
    v16h b = *(const v16h*)(brow + k0 + kb);
    acc = __builtin_amdgcn_wmma_f32_16x16x32_f16(false, a, false, b,
                                                 (short)0, acc, false, false);
  }
  int n = nt * 16 + r;
  float bv = 0.0f;
  if (bias1) bv += bias1[n];
  if (bias2) bv += bias2[n];
#pragma unroll
  for (int i = 0; i < 8; ++i) {
    int m = mt * 16 + hi * 8 + i;
    float val = acc[i] + bv;
    if (relu) val = fmaxf(val, 0.0f);
    if (Cf) Cf[(size_t)m * ldc + n] = val;
    if (Ch) Ch[(size_t)m * ldc + n] = (_Float16)val;
  }
}

// ---------------------------------------------------------------------------
// hp = (shift(h_seq) @ W_proj.T) * mask       [8192 x 128]
// ---------------------------------------------------------------------------
__global__ void hp_gemm_kernel(const _Float16* __restrict__ h_seq_h,
                               const _Float16* __restrict__ Wproj_h,
                               const float* __restrict__ visit_mask,
                               float* __restrict__ hp)
{
  int w = blockIdx.x * (blockDim.x >> 5) + (threadIdx.x >> 5);
  if (w >= 512 * 8) return;
  int nt = w & 7, mt = w >> 3;
  int lane = threadIdx.x & 31;
  int r = lane & 15, hi = lane >> 4;
  int mrow = mt * 16 + r;
  bool valid = (mrow & (L_ - 1)) != 0;   // l==0 rows use zero h_prev
  const _Float16* arow = valid ? (h_seq_h + (size_t)(mrow - 1) * H_) : h_seq_h;
  const _Float16* brow = Wproj_h + (size_t)(nt * 16 + r) * H_;
  const int ka = hi * 16, kb = hi * 16;
  v8f acc = {};
  for (int k0 = 0; k0 < H_; k0 += 32) {
    v16h a;
    if (valid) {
      a = load_a16(arow + k0 + ka);
    } else {
#pragma unroll
      for (int i = 0; i < 16; ++i) a[i] = (_Float16)0.0f;
    }
    v16h b = *(const v16h*)(brow + k0 + kb);
    acc = __builtin_amdgcn_wmma_f32_16x16x32_f16(false, a, false, b,
                                                 (short)0, acc, false, false);
  }
  int n = nt * 16 + r;
#pragma unroll
  for (int i = 0; i < 8; ++i) {
    int m = mt * 16 + hi * 8 + i;
    hp[(size_t)m * E_ + n] = acc[i] * visit_mask[m];
  }
}

// ---------------------------------------------------------------------------
// One LSTM timestep: 4 fused gate GEMMs (h_prev @ W_hh.T) + nonlinearity.
// h-carry kept in f16 (GEMM A operand); c-state kept in f32.
// ---------------------------------------------------------------------------
__global__ void lstm_step_kernel(const float* __restrict__ xg,
                                 const _Float16* __restrict__ Whh_h,
                                 const float* __restrict__ visit_mask,
                                 const _Float16* __restrict__ h_prev,
                                 _Float16* __restrict__ h_next,
                                 float* __restrict__ c_car,
                                 float* __restrict__ h_seq,
                                 _Float16* __restrict__ h_seq_h, int l)
{
  int w = blockIdx.x * (blockDim.x >> 5) + (threadIdx.x >> 5);
  if (w >= 64) return;
  int nt = w & 7, mt = w >> 3;
  int lane = threadIdx.x & 31;
  int r = lane & 15, hi = lane >> 4;
  v8f a0 = {}, a1 = {}, a2 = {}, a3 = {};
  if (l > 0) {
    const _Float16* arow  = h_prev + (size_t)(mt * 16 + r) * H_;
    const _Float16* bbase = Whh_h + (size_t)(nt * 16 + r) * H_;
    const int ka = hi * 16, kb = hi * 16;
    for (int k0 = 0; k0 < H_; k0 += 32) {
      v16h a = load_a16(arow + k0 + ka);
      const _Float16* bp = bbase + k0 + kb;
      v16h b0 = *(const v16h*)(bp);
      v16h b1 = *(const v16h*)(bp + H_ * H_);
      v16h b2 = *(const v16h*)(bp + 2 * H_ * H_);
      v16h b3 = *(const v16h*)(bp + 3 * H_ * H_);
      a0 = __builtin_amdgcn_wmma_f32_16x16x32_f16(false, a, false, b0, (short)0, a0, false, false);
      a1 = __builtin_amdgcn_wmma_f32_16x16x32_f16(false, a, false, b1, (short)0, a1, false, false);
      a2 = __builtin_amdgcn_wmma_f32_16x16x32_f16(false, a, false, b2, (short)0, a2, false, false);
      a3 = __builtin_amdgcn_wmma_f32_16x16x32_f16(false, a, false, b3, (short)0, a3, false, false);
    }
  }
  int n = nt * 16 + r;
#pragma unroll
  for (int i = 0; i < 8; ++i) {
    int m = mt * 16 + hi * 8 + i;                 // batch index
    size_t xrow = ((size_t)m * L_ + l) * (4 * H_);
    float gi = a0[i] + xg[xrow + n];
    float gf = a1[i] + xg[xrow + H_ + n];
    float gg = a2[i] + xg[xrow + 2 * H_ + n];
    float go = a3[i] + xg[xrow + 3 * H_ + n];
    float si = sigmoidf_(gi);
    float sf = sigmoidf_(gf);
    float tg = tanhf(gg);
    float so = sigmoidf_(go);
    float cp = (l == 0) ? 0.0f : c_car[m * H_ + n];
    float cn = sf * cp + si * tg;
    float hn = so * tanhf(cn);
    c_car[m * H_ + n] = cn;
    h_next[m * H_ + n] = (_Float16)hn;            // un-masked carry (f16)
    float hm = hn * visit_mask[m * L_ + l];
    h_seq[((size_t)m * L_ + l) * H_ + n]   = hm;
    h_seq_h[((size_t)m * L_ + l) * H_ + n] = (_Float16)hm;
  }
}

// ---------------------------------------------------------------------------
// last-hidden gather + classifier logit (one block per batch row)
// ---------------------------------------------------------------------------
__global__ void logits_kernel(const float* __restrict__ h_seq,
                              const float* __restrict__ visit_mask,
                              const float* __restrict__ W_cls,
                              const float* __restrict__ b_cls,
                              float* __restrict__ outp)
{
  __shared__ float red[128];
  __shared__ int len_s;
  int b = blockIdx.x;
  int t = threadIdx.x;
  red[t] = (t < L_) ? visit_mask[b * L_ + t] : 0.0f;
  __syncthreads();
  for (int k = 64; k > 0; k >>= 1) { if (t < k) red[t] += red[t + k]; __syncthreads(); }
  if (t == 0) { int len = (int)(red[0] + 0.5f); len_s = len < 1 ? 1 : len; }
  __syncthreads();
  int len = len_s;
  float p = h_seq[((size_t)b * L_ + (len - 1)) * H_ + t] * W_cls[t];
  red[t] = p;
  __syncthreads();
  for (int k = 64; k > 0; k >>= 1) { if (t < k) red[t] += red[t + k]; __syncthreads(); }
  if (t == 0) outp[b] = red[0] + b_cls[0];
}

// ---------------------------------------------------------------------------
// sum(mask) -> scal[0]; zero the loss accumulator
// ---------------------------------------------------------------------------
__global__ void masksum_kernel(const float* __restrict__ visit_mask,
                               float* __restrict__ scal,
                               float* __restrict__ loss_out)
{
  __shared__ float red[256];
  int t = threadIdx.x;
  float s = 0.0f;
  for (int i = t; i < ROWS_; i += 256) s += visit_mask[i];
  red[t] = s; __syncthreads();
  for (int k = 128; k > 0; k >>= 1) { if (t < k) red[t] += red[t + k]; __syncthreads(); }
  if (t == 0) { scal[0] = red[0] < 1.0f ? 1.0f : red[0]; loss_out[0] = 0.0f; }
}

// ---------------------------------------------------------------------------
// z_t = (sqrt(ab)*v + sqrt(1-ab)*eps) * mask
// ---------------------------------------------------------------------------
__global__ void zt_kernel(const float* __restrict__ v, const float* __restrict__ eps,
                          const float* __restrict__ visit_mask,
                          const int* __restrict__ t_diff, float* __restrict__ zt)
{
  int idx = blockIdx.x * blockDim.x + threadIdx.x;
  if (idx >= ROWS_ * E_) return;
  int m = idx >> 7;
  float mask = visit_mask[m];
  int t = (mask > 0.0f) ? t_diff[m >> 6] : 1;
  float ab = 1.0f;
  for (int i = 0; i < t; ++i) ab *= (1.0f - beta_at(i));
  zt[idx] = (sqrtf(ab) * v[idx] + sqrtf(1.0f - ab) * eps[idx]) * mask;
}

// ---------------------------------------------------------------------------
// fuse gate: 2-way softmax over [z,hp]@W_fuse.T+b; xcat_h = [g, time_embed[t]] (f16)
// kk>0: sampling step kk; kk==0: use t_diff[b].
// ---------------------------------------------------------------------------
__global__ void fuse_gate_kernel(const float* __restrict__ z,
                                 const float* __restrict__ hp,
                                 const float* __restrict__ visit_mask,
                                 const int* __restrict__ t_diff,
                                 const float* __restrict__ W_fuse,
                                 const float* __restrict__ b_fuse,
                                 const float* __restrict__ time_embed,
                                 _Float16* __restrict__ xcat_h, int kk)
{
  __shared__ float r0[128], r1[128];
  int row = blockIdx.x;
  int t = threadIdx.x;
  float zv = z[(size_t)row * E_ + t];
  float hv = hp[(size_t)row * E_ + t];
  r0[t] = zv * W_fuse[t]          + hv * W_fuse[E_ + t];
  r1[t] = zv * W_fuse[2 * E_ + t] + hv * W_fuse[3 * E_ + t];
  __syncthreads();
  for (int k = 64; k > 0; k >>= 1) {
    if (t < k) { r0[t] += r0[t + k]; r1[t] += r1[t + k]; }
    __syncthreads();
  }
  float a0 = r0[0] + b_fuse[0];
  float a1 = r1[0] + b_fuse[1];
  float mx = fmaxf(a0, a1);
  float e0 = __expf(a0 - mx), e1 = __expf(a1 - mx);
  float inv = 1.0f / (e0 + e1);
  a0 = e0 * inv; a1 = e1 * inv;
  float mask = visit_mask[row];
  int tt = (mask > 0.0f) ? (kk > 0 ? kk : t_diff[row >> 6]) : 1;
  xcat_h[(size_t)row * (2 * E_) + t]      = (_Float16)(a0 * zv + a1 * hv);
  xcat_h[(size_t)row * (2 * E_) + E_ + t] = (_Float16)time_embed[(size_t)tt * E_ + t];
}

// ---------------------------------------------------------------------------
// diffusion loss: atomicAdd( sum((eps_pred-eps)^2 * mask) / denom )
// ---------------------------------------------------------------------------
__global__ void loss_kernel(const float* __restrict__ epsp, const float* __restrict__ eps,
                            const float* __restrict__ visit_mask,
                            const float* __restrict__ scal, float* __restrict__ outp)
{
  __shared__ float red[256];
  int t = threadIdx.x;
  float s = 0.0f;
  for (size_t i = (size_t)blockIdx.x * 256 + t; i < (size_t)ROWS_ * E_;
       i += (size_t)gridDim.x * 256) {
    float d = epsp[i] - eps[i];
    s += d * d * visit_mask[i >> 7];
  }
  red[t] = s; __syncthreads();
  for (int k = 128; k > 0; k >>= 1) { if (t < k) red[t] += red[t + k]; __syncthreads(); }
  if (t == 0) atomicAdd(outp, red[0] / scal[0]);
}

// ---------------------------------------------------------------------------
// reverse diffusion update
// ---------------------------------------------------------------------------
__global__ void sample_update_kernel(float* __restrict__ z,
                                     const float* __restrict__ epsp,
                                     const float* __restrict__ noise,
                                     const float* __restrict__ visit_mask,
                                     int kk)
{
  int idx = blockIdx.x * blockDim.x + threadIdx.x;
  if (idx >= ROWS_ * E_) return;
  int m = idx >> 7;
  float ab = 1.0f;
  for (int i = 0; i < kk; ++i) ab *= (1.0f - beta_at(i));
  float beta = beta_at(kk - 1);
  float alpha = 1.0f - beta;
  float mean = (z[idx] - beta * rsqrtf(1.0f - ab) * epsp[idx]) * rsqrtf(alpha);
  float zn = (kk > 1) ? (mean + sqrtf(beta) * noise[idx]) : mean;
  z[idx] = zn * visit_mask[m];
}

// ---------------------------------------------------------------------------
extern "C" void kernel_launch(void* const* d_in, const int* in_sizes, int n_in,
                              void* d_out, int out_size, void* d_ws, size_t ws_size,
                              hipStream_t stream)
{
  const float* padded_x    = (const float*)d_in[0];
  const float* visit_mask  = (const float*)d_in[1];
  const int*   padded_bins = (const int*)  d_in[2];
  const int*   t_diff      = (const int*)  d_in[3];
  const float* eps         = (const float*)d_in[4];
  const float* z0          = (const float*)d_in[5];
  const float* noise       = (const float*)d_in[6];
  const float* W_visit     = (const float*)d_in[7];
  const float* bin_embed   = (const float*)d_in[8];
  const float* W_ih        = (const float*)d_in[9];
  const float* W_hh        = (const float*)d_in[10];
  const float* b_ih        = (const float*)d_in[11];
  const float* b_hh        = (const float*)d_in[12];
  const float* W_cls       = (const float*)d_in[13];
  const float* b_cls       = (const float*)d_in[14];
  const float* W_proj      = (const float*)d_in[15];
  const float* time_embed  = (const float*)d_in[16];
  const float* W_fuse      = (const float*)d_in[17];
  const float* b_fuse      = (const float*)d_in[18];
  const float* W_e1        = (const float*)d_in[19];
  const float* b_e1        = (const float*)d_in[20];
  const float* W_e2        = (const float*)d_in[21];
  const float* b_e2        = (const float*)d_in[22];

  float* out = (float*)d_out;          // [0..127]=real, [128..255]=syn, [256]=loss

  // ---- workspace layout (float units; f16 buffers counted as n/2 floats) ----
  float* ws       = (float*)d_ws;
  float*     v        = ws;                         // 1,048,576 f
  float*     xg       = v + 1048576;                // 4,194,304 f
  float*     h_seq    = xg + 4194304;               // 1,048,576 f
  float*     c_car    = h_seq + 1048576;            // 16,384 f
  float*     hp       = c_car + 16384;              // 1,048,576 f
  float*     zbuf     = hp + 1048576;               // 1,048,576 f
  float*     epsp     = zbuf + 1048576;             // 1,048,576 f
  float*     scal     = epsp + 1048576;             // 16 f (aligned pad)
  _Float16*  v_h      = (_Float16*)(scal + 16);     // 1,048,576 h
  _Float16*  h_seq_h  = v_h + 1048576;              // 1,048,576 h
  _Float16*  hc16A    = h_seq_h + 1048576;          // 16,384 h
  _Float16*  hc16B    = hc16A + 16384;              // 16,384 h
  _Float16*  xcat_h   = hc16B + 16384;              // 2,097,152 h
  _Float16*  y1_h     = xcat_h + 2097152;           // 1,048,576 h
  _Float16*  zbuf_h   = y1_h + 1048576;             // 1,048,576 h
  _Float16*  Wv_h     = zbuf_h + 1048576;           // 1,048,576 h
  _Float16*  Wih_h    = Wv_h + 1048576;             // 65,536 h
  _Float16*  Whh_h    = Wih_h + 65536;              // 65,536 h
  _Float16*  Wproj_h  = Whh_h + 65536;              // 16,384 h
  _Float16*  We1_h    = Wproj_h + 16384;            // 32,768 h
  _Float16*  We2_h    = We1_h + 32768;              // 16,384 h

  // 0) one-time weight conversions to f16
  cvt16_kernel<<<4096, 256, 0, stream>>>(W_visit, Wv_h, V_ * E_);
  cvt16_kernel<<<256, 256, 0, stream>>>(W_ih, Wih_h, 4 * H_ * E_);
  cvt16_kernel<<<256, 256, 0, stream>>>(W_hh, Whh_h, 4 * H_ * H_);
  cvt16_kernel<<<64, 256, 0, stream>>>(W_proj, Wproj_h, E_ * H_);
  cvt16_kernel<<<128, 256, 0, stream>>>(W_e1, We1_h, E_ * 2 * E_);
  cvt16_kernel<<<64, 256, 0, stream>>>(W_e2, We2_h, E_ * E_);

  // 1) visit embedding (memory-bound 256MB stream of padded_x)
  visit_embed_kernel<<<512, 256, 0, stream>>>(padded_x, Wv_h, padded_bins,
                                              visit_mask, bin_embed, v, v_h);
  // 2) mask sum + zero loss accumulator
  masksum_kernel<<<1, 256, 0, stream>>>(visit_mask, scal, out + 256);

  auto run_lstm = [&](const _Float16* vin_h) {
    // xg = vin @ W_ih.T + b_ih + b_hh : M=8192 K=128 N=512 (512x32 tiles)
    gemm_h_kernel<<<2048, 256, 0, stream>>>(vin_h, 128, Wih_h, 128, xg, nullptr,
                                            512, 512, 32, b_ih, b_hh, 0);
    for (int l = 0; l < L_; ++l) {
      _Float16* hprev = (l & 1) ? hc16B : hc16A;
      _Float16* hnext = (l & 1) ? hc16A : hc16B;
      lstm_step_kernel<<<8, 256, 0, stream>>>(xg, Whh_h, visit_mask, hprev, hnext,
                                              c_car, h_seq, h_seq_h, l);
    }
  };

  // 3) encoder LSTM over real visits + real logits
  run_lstm(v_h);
  logits_kernel<<<128, 128, 0, stream>>>(h_seq, visit_mask, W_cls, b_cls, out);

  // 4) hp = shift(h) @ W_proj.T * mask
  hp_gemm_kernel<<<512, 256, 0, stream>>>(h_seq_h, Wproj_h, visit_mask, hp);

  // 5) diffusion loss branch
  zt_kernel<<<4096, 256, 0, stream>>>(v, eps, visit_mask, t_diff, zbuf);
  fuse_gate_kernel<<<8192, 128, 0, stream>>>(zbuf, hp, visit_mask, t_diff,
                                             W_fuse, b_fuse, time_embed, xcat_h, 0);
  gemm_h_kernel<<<512, 256, 0, stream>>>(xcat_h, 256, We1_h, 256, nullptr, y1_h,
                                         128, 512, 8, b_e1, nullptr, 1);
  gemm_h_kernel<<<512, 256, 0, stream>>>(y1_h, 128, We2_h, 128, epsp, nullptr,
                                         128, 512, 8, b_e2, nullptr, 0);
  loss_kernel<<<64, 256, 0, stream>>>(epsp, eps, visit_mask, scal, out + 256);

  // 6) reverse diffusion sampling
  hipMemcpyAsync(zbuf, z0, (size_t)ROWS_ * E_ * sizeof(float),
                 hipMemcpyDeviceToDevice, stream);
  for (int kk = STEPS_; kk >= 1; --kk) {
    fuse_gate_kernel<<<8192, 128, 0, stream>>>(zbuf, hp, visit_mask, t_diff,
                                               W_fuse, b_fuse, time_embed, xcat_h, kk);
    gemm_h_kernel<<<512, 256, 0, stream>>>(xcat_h, 256, We1_h, 256, nullptr, y1_h,
                                           128, 512, 8, b_e1, nullptr, 1);
    gemm_h_kernel<<<512, 256, 0, stream>>>(y1_h, 128, We2_h, 128, epsp, nullptr,
                                           128, 512, 8, b_e2, nullptr, 0);
    sample_update_kernel<<<4096, 256, 0, stream>>>(
        zbuf, epsp, noise + (size_t)(STEPS_ - kk) * ROWS_ * E_, visit_mask, kk);
  }

  // 7) encoder LSTM over synthetic visits + synthetic logits
  cvt16_kernel<<<4096, 256, 0, stream>>>(zbuf, zbuf_h, ROWS_ * E_);
  run_lstm(zbuf_h);
  logits_kernel<<<128, 128, 0, stream>>>(h_seq, visit_mask, W_cls, b_cls, out + 128);
}
